// QuantumAttention_76459007803724
// MI455X (gfx1250) — matrix operand order repaired and checked by reading, not compiled, not run on hardware
//
#include <hip/hip_runtime.h>
#include <cstdint>
#include <cstddef>

#define B_  2
#define S_  2048
#define H_  1024
#define NH_ 16
#define DH_ 64

typedef __attribute__((ext_vector_type(16))) _Float16 v16h;
typedef __attribute__((ext_vector_type(8)))  _Float16 v8h;
typedef __attribute__((ext_vector_type(8)))  float    v8f;

__device__ __forceinline__ v8f wmma32f16(v16h a, v16h b, v8f c) {
  return __builtin_amdgcn_wmma_f32_16x16x32_f16(false, a, false, b, (short)0, c, false, false);
}

// A fragment (16x32 f16) built from an fp32 row-major row; K window [d0, d0+32).
// Lane layout: halves 0..7 -> K = d0 + h + 8g ; halves 8..15 -> K = d0 + 16 + (h-8) + 8g
__device__ __forceinline__ v16h a_frag_f32(const float* rowp, int d0, int g) {
  const float4* pa = reinterpret_cast<const float4*>(rowp + d0 + 8 * g);
  const float4* pb = reinterpret_cast<const float4*>(rowp + d0 + 16 + 8 * g);
  float4 x0 = pa[0], x1 = pa[1], y0 = pb[0], y1 = pb[1];
  v16h a;
  a[0]=(_Float16)x0.x; a[1]=(_Float16)x0.y; a[2]=(_Float16)x0.z; a[3]=(_Float16)x0.w;
  a[4]=(_Float16)x1.x; a[5]=(_Float16)x1.y; a[6]=(_Float16)x1.z; a[7]=(_Float16)x1.w;
  a[8]=(_Float16)y0.x; a[9]=(_Float16)y0.y; a[10]=(_Float16)y0.z; a[11]=(_Float16)y0.w;
  a[12]=(_Float16)y1.x; a[13]=(_Float16)y1.y; a[14]=(_Float16)y1.z; a[15]=(_Float16)y1.w;
  return a;
}

// Same A fragment from a row-major f16 row (global or LDS).
__device__ __forceinline__ v16h a_frag_f16(const _Float16* rowp, int d0, int g) {
  v8h lo = *reinterpret_cast<const v8h*>(rowp + d0 + 8 * g);
  v8h hi = *reinterpret_cast<const v8h*>(rowp + d0 + 16 + 8 * g);
  v16h a;
#pragma unroll
  for (int i = 0; i < 8; ++i) { a[i] = lo[i]; a[i + 8] = hi[i]; }
  return a;
}

// B fragment (32x16 f16) from a "transposed" f16 matrix WT[n][k] (kdim = row length).
// Lane holds column n = n0 + (lane&15); halves 0..7 -> k = k0 + 8*(lane>>4) + h ; halves 8..15 -> k = k0 + 16 + 8*(lane>>4) + (h-8).
__device__ __forceinline__ v16h b_frag(const _Float16* WT, int kdim, int n0, int k0, int lane) {
  const _Float16* p = WT + (size_t)(n0 + (lane & 15)) * kdim + k0 + 8 * (lane >> 4);
  v8h lo = *reinterpret_cast<const v8h*>(p);
  v8h hi = *reinterpret_cast<const v8h*>(p + 16);
  v16h b;
#pragma unroll
  for (int i = 0; i < 8; ++i) { b[i] = lo[i]; b[i + 8] = hi[i]; }
  return b;
}

// ---------------------------------------------------------------------------
// Kernel 0: pack / transpose weights to f16
// ---------------------------------------------------------------------------
__global__ __launch_bounds__(256) void k_pack(
    const float* __restrict__ Wq, const float* __restrict__ Wk, const float* __restrict__ Wv,
    const float* __restrict__ Uqr, const float* __restrict__ Uqi,
    const float* __restrict__ Ukr, const float* __restrict__ Uki,
    const float* __restrict__ Uvr, const float* __restrict__ Uvi,
    const float* __restrict__ Wo,
    _Float16* __restrict__ WqT, _Float16* __restrict__ WkT, _Float16* __restrict__ WvT,
    _Float16* __restrict__ UqrT, _Float16* __restrict__ UqiT,
    _Float16* __restrict__ UkrT, _Float16* __restrict__ UkiT,
    _Float16* __restrict__ UvrT, _Float16* __restrict__ UviT,
    _Float16* __restrict__ WoT)
{
  const int j = blockIdx.x * 256 + threadIdx.x;   // grid covers H_*H_ exactly
  {
    const int n = j / H_, k = j % H_;
    WoT[j] = (_Float16)Wo[(size_t)k * H_ + n];    // WoT[n][k] = Wo[k][n]
  }
  if (j < 2 * DH_ * DH_) {                        // 128 x 64
    const int n = j / DH_, k = j % DH_;
    WqT[j] = (_Float16)Wq[k * (2 * DH_) + n];
    WkT[j] = (_Float16)Wk[k * (2 * DH_) + n];
    WvT[j] = (_Float16)Wv[k * (2 * DH_) + n];
  }
  if (j < DH_ * DH_) {                            // 64 x 64
    const int n = j / DH_, k = j % DH_;
    const int src = k * DH_ + n;
    UqrT[j] = (_Float16)Uqr[src]; UqiT[j] = (_Float16)Uqi[src];
    UkrT[j] = (_Float16)Ukr[src]; UkiT[j] = (_Float16)Uki[src];
    UvrT[j] = (_Float16)Uvr[src]; UviT[j] = (_Float16)Uvi[src];
  }
}

// ---------------------------------------------------------------------------
// Kernel 1: state prep (linear + unit-normalize) + quantum gate, per 16 tokens
// ---------------------------------------------------------------------------
__global__ __launch_bounds__(32) void k_prep_gate(
    const float* __restrict__ hidden,
    const float* __restrict__ bq, const float* __restrict__ bk, const float* __restrict__ bv,
    const _Float16* __restrict__ WqT, const _Float16* __restrict__ WkT, const _Float16* __restrict__ WvT,
    const _Float16* __restrict__ UqrT, const _Float16* __restrict__ UqiT,
    const _Float16* __restrict__ UkrT, const _Float16* __restrict__ UkiT,
    const _Float16* __restrict__ UvrT, const _Float16* __restrict__ UviT,
    _Float16* __restrict__ qr, _Float16* __restrict__ qi,
    _Float16* __restrict__ kr, _Float16* __restrict__ ki,
    _Float16* __restrict__ vmagT)
{
  const int lane = threadIdx.x;
  const int g = lane >> 4, nn = lane & 15;
  const int tiles = S_ / 16;
  const int bh = blockIdx.x / tiles;
  const int s0 = (blockIdx.x % tiles) * 16;
  const int b = bh / NH_, hd = bh % NH_;

  const float* xrow = hidden + ((size_t)(b * S_ + s0 + nn)) * H_ + hd * DH_;
  const v16h Ax0 = a_frag_f32(xrow, 0, g);
  const v16h Ax1 = a_frag_f32(xrow, 32, g);

  __shared__ alignas(16) _Float16 lr[16][72];
  __shared__ alignas(16) _Float16 li[16][72];

  for (int proj = 0; proj < 3; ++proj) {
    const _Float16* WT  = proj == 0 ? WqT  : proj == 1 ? WkT  : WvT;
    const float*   bias = proj == 0 ? bq   : proj == 1 ? bk   : bv;
    const _Float16* UrT = proj == 0 ? UqrT : proj == 1 ? UkrT : UvrT;
    const _Float16* UiT = proj == 0 ? UqiT : proj == 1 ? UkiT : UviT;

    // c = x @ W + b  ([16,64]@[64,128]); pair columns (j, j+64) -> normalize
#pragma unroll
    for (int ct = 0; ct < 4; ++ct) {
      v8f cr = {}, ci = {};
      cr = wmma32f16(Ax0, b_frag(WT, DH_, ct * 16, 0, lane), cr);
      cr = wmma32f16(Ax1, b_frag(WT, DH_, ct * 16, 32, lane), cr);
      ci = wmma32f16(Ax0, b_frag(WT, DH_, 64 + ct * 16, 0, lane), ci);
      ci = wmma32f16(Ax1, b_frag(WT, DH_, 64 + ct * 16, 32, lane), ci);
      const float br = bias[ct * 16 + nn];
      const float bi = bias[64 + ct * 16 + nn];
#pragma unroll
      for (int r = 0; r < 8; ++r) {
        const float rr = cr[r] + br, ii = ci[r] + bi;
        const float inv = 1.0f / sqrtf(rr * rr + ii * ii + 1e-12f);
        lr[r + 8 * g][ct * 16 + nn] = (_Float16)(rr * inv);
        li[r + 8 * g][ct * 16 + nn] = (_Float16)(ii * inv);
      }
    }
    __syncthreads();

    // gate: (r + j i) @ (Ur + j Ui)
    const _Float16* lrrow = &lr[nn][0];
    const _Float16* lirow = &li[nn][0];
    const v16h Ar0 = a_frag_f16(lrrow, 0, g), Ar1 = a_frag_f16(lrrow, 32, g);
    const v16h Ai0 = a_frag_f16(lirow, 0, g), Ai1 = a_frag_f16(lirow, 32, g);
    const v16h nAi0 = -Ai0, nAi1 = -Ai1;   // f16 WMMA has no A/B neg -> fold here

#pragma unroll
    for (int ct = 0; ct < 4; ++ct) {
      v8f dr = {}, di = {};
      const v16h Bur0 = b_frag(UrT, DH_, ct * 16, 0, lane);
      const v16h Bur1 = b_frag(UrT, DH_, ct * 16, 32, lane);
      const v16h Bui0 = b_frag(UiT, DH_, ct * 16, 0, lane);
      const v16h Bui1 = b_frag(UiT, DH_, ct * 16, 32, lane);
      dr = wmma32f16(Ar0, Bur0, dr); dr = wmma32f16(Ar1, Bur1, dr);
      dr = wmma32f16(nAi0, Bui0, dr); dr = wmma32f16(nAi1, Bui1, dr);
      di = wmma32f16(Ar0, Bui0, di); di = wmma32f16(Ar1, Bui1, di);
      di = wmma32f16(Ai0, Bur0, di); di = wmma32f16(Ai1, Bur1, di);

      if (proj == 2) {
        v8h vm;
#pragma unroll
        for (int r = 0; r < 8; ++r)
          vm[r] = (_Float16)sqrtf(dr[r] * dr[r] + di[r] * di[r]);
        // |v| stored transposed: vmagT[bh][d][k_seq]
        *reinterpret_cast<v8h*>(vmagT + (size_t)bh * DH_ * S_ +
                                (size_t)(ct * 16 + nn) * S_ + s0 + 8 * g) = vm;
      } else {
        _Float16* orp = (proj == 0) ? qr : kr;
        _Float16* oip = (proj == 0) ? qi : ki;
#pragma unroll
        for (int r = 0; r < 8; ++r) {
          const size_t idx = ((size_t)bh * S_ + s0 + r + 8 * g) * DH_ + ct * 16 + nn;
          orp[idx] = (_Float16)dr[r];
          oip[idx] = (_Float16)di[r];
        }
      }
    }
    __syncthreads();
  }
}

// ---------------------------------------------------------------------------
// Kernel 2: complex scores -> |s|/8, threshold, write to probs region
// one wave: 16 queries x 128 keys
// ---------------------------------------------------------------------------
__global__ __launch_bounds__(32) void k_scores(
    const _Float16* __restrict__ qr, const _Float16* __restrict__ qi,
    const _Float16* __restrict__ kr, const _Float16* __restrict__ ki,
    float* __restrict__ probs)
{
  const int lane = threadIdx.x;
  const int g = lane >> 4, nn = lane & 15;
  const int kgroups = S_ / 128;  // 16
  const int qtiles  = S_ / 16;   // 128
  int idx = blockIdx.x;
  const int kg = idx % kgroups; idx /= kgroups;
  const int qt = idx % qtiles;  idx /= qtiles;
  const int bh = idx;

  const _Float16* qrrow = qr + ((size_t)bh * S_ + qt * 16 + nn) * DH_;
  const _Float16* qirow = qi + ((size_t)bh * S_ + qt * 16 + nn) * DH_;
  const v16h Aqr0 = a_frag_f16(qrrow, 0, g), Aqr1 = a_frag_f16(qrrow, 32, g);
  const v16h Aqi0 = a_frag_f16(qirow, 0, g), Aqi1 = a_frag_f16(qirow, 32, g);
  const v16h nAqi0 = -Aqi0, nAqi1 = -Aqi1;

  const _Float16* krb = kr + (size_t)bh * S_ * DH_;
  const _Float16* kib = ki + (size_t)bh * S_ * DH_;
  float* prow = probs + ((size_t)bh * S_ + qt * 16) * S_ + (size_t)kg * 128;

#pragma unroll
  for (int kt = 0; kt < 8; ++kt) {
    const int key0 = kg * 128 + kt * 16;
    const v16h Bkr0 = b_frag(krb, DH_, key0, 0, lane);
    const v16h Bkr1 = b_frag(krb, DH_, key0, 32, lane);
    const v16h Bki0 = b_frag(kib, DH_, key0, 0, lane);
    const v16h Bki1 = b_frag(kib, DH_, key0, 32, lane);
    v8f sr = {}, si = {};
    sr = wmma32f16(Aqr0, Bkr0, sr);  sr = wmma32f16(Aqr1, Bkr1, sr);
    sr = wmma32f16(nAqi0, Bki0, sr); sr = wmma32f16(nAqi1, Bki1, sr);
    si = wmma32f16(Aqr0, Bki0, si);  si = wmma32f16(Aqr1, Bki1, si);
    si = wmma32f16(Aqi0, Bkr0, si);  si = wmma32f16(Aqi1, Bkr1, si);
#pragma unroll
    for (int r = 0; r < 8; ++r) {
      float m = sqrtf(sr[r] * sr[r] + si[r] * si[r]) * 0.125f;  // SCALE = 1/sqrt(64)
      m = (m < 0.1f) ? 0.0f : m;                                // noise threshold
      prow[(size_t)(r + 8 * g) * S_ + kt * 16 + nn] = m;
    }
  }
}

// ---------------------------------------------------------------------------
// Kernel 3: in-place row softmax over 2048 elements
// ---------------------------------------------------------------------------
__global__ __launch_bounds__(256) void k_softmax(float* __restrict__ probs) {
  const int tid = threadIdx.x;
  float* p = probs + (size_t)blockIdx.x * S_;
  float4 a = reinterpret_cast<float4*>(p)[tid * 2];
  float4 b = reinterpret_cast<float4*>(p)[tid * 2 + 1];
  __shared__ float red[256];

  float mx = fmaxf(fmaxf(fmaxf(a.x, a.y), fmaxf(a.z, a.w)),
                   fmaxf(fmaxf(b.x, b.y), fmaxf(b.z, b.w)));
  red[tid] = mx; __syncthreads();
  for (int s = 128; s > 0; s >>= 1) { if (tid < s) red[tid] = fmaxf(red[tid], red[tid + s]); __syncthreads(); }
  mx = red[0]; __syncthreads();

  a.x = __expf(a.x - mx); a.y = __expf(a.y - mx); a.z = __expf(a.z - mx); a.w = __expf(a.w - mx);
  b.x = __expf(b.x - mx); b.y = __expf(b.y - mx); b.z = __expf(b.z - mx); b.w = __expf(b.w - mx);
  red[tid] = a.x + a.y + a.z + a.w + b.x + b.y + b.z + b.w; __syncthreads();
  for (int s = 128; s > 0; s >>= 1) { if (tid < s) red[tid] += red[tid + s]; __syncthreads(); }
  const float inv = 1.0f / red[0];

  a.x *= inv; a.y *= inv; a.z *= inv; a.w *= inv;
  b.x *= inv; b.y *= inv; b.z *= inv; b.w *= inv;
  reinterpret_cast<float4*>(p)[tid * 2] = a;
  reinterpret_cast<float4*>(p)[tid * 2 + 1] = b;
}

// ---------------------------------------------------------------------------
// Kernel 4: ctx = probs @ |V|   ([2048,2048]@[2048,64] per head)
// ---------------------------------------------------------------------------
__global__ __launch_bounds__(32) void k_ctx(
    const float* __restrict__ probs, const _Float16* __restrict__ vmagT,
    _Float16* __restrict__ ctx)
{
  const int lane = threadIdx.x;
  const int g = lane >> 4, nn = lane & 15;
  const int qtiles = S_ / 16;
  const int bh = blockIdx.x / qtiles;
  const int qt = blockIdx.x % qtiles;
  const int b = bh / NH_, hd = bh % NH_;

  const float* prow = probs + ((size_t)bh * S_ + qt * 16 + nn) * S_;
  const _Float16* vb = vmagT + (size_t)bh * DH_ * S_;

  v8f acc[4] = {};
  for (int ks = 0; ks < S_ / 32; ++ks) {
    const int k0 = ks * 32;
    const v16h Af = a_frag_f32(prow, k0, g);   // fp32 probs -> f16 A fragment
#pragma unroll
    for (int ct = 0; ct < 4; ++ct)
      acc[ct] = wmma32f16(Af, b_frag(vb, S_, ct * 16, k0, lane), acc[ct]);
  }

  __shared__ alignas(16) _Float16 tile[16][72];
#pragma unroll
  for (int ct = 0; ct < 4; ++ct)
#pragma unroll
    for (int r = 0; r < 8; ++r)
      tile[r + 8 * g][ct * 16 + nn] = (_Float16)acc[ct][r];
  __syncthreads();

#pragma unroll
  for (int i = 0; i < 4; ++i) {
    const int c = lane + 32 * i;       // 128 chunks of 8 halves = 16x64 tile
    const int r = c >> 3, seg = c & 7;
    const v8h val = *reinterpret_cast<const v8h*>(&tile[r][seg * 8]);
    *reinterpret_cast<v8h*>(ctx + ((size_t)(b * S_ + qt * 16 + r)) * H_ +
                            hd * DH_ + seg * 8) = val;
  }
}

// ---------------------------------------------------------------------------
// Kernel 5: out = ctx @ Wo + bo + hidden (residual), fp32 result
// ---------------------------------------------------------------------------
__global__ __launch_bounds__(32) void k_outproj(
    const _Float16* __restrict__ ctx, const _Float16* __restrict__ WoT,
    const float* __restrict__ bo, const float* __restrict__ hidden,
    float* __restrict__ res)
{
  const int lane = threadIdx.x;
  const int g = lane >> 4, nn = lane & 15;
  const int ngroups = H_ / 64;   // 16
  const int rt = blockIdx.x / ngroups;
  const int ng = blockIdx.x % ngroups;

  const _Float16* arow = ctx + (size_t)(rt * 16 + nn) * H_;
  v8f acc[4] = {};
  for (int ks = 0; ks < H_ / 32; ++ks) {
    const int k0 = ks * 32;
    const v16h Af = a_frag_f16(arow, k0, g);
#pragma unroll
    for (int t = 0; t < 4; ++t)
      acc[t] = wmma32f16(Af, b_frag(WoT, H_, ng * 64 + t * 16, k0, lane), acc[t]);
  }

#pragma unroll
  for (int t = 0; t < 4; ++t) {
    const int col = ng * 64 + t * 16 + nn;
    const float bias = bo[col];
#pragma unroll
    for (int r = 0; r < 8; ++r) {
      const size_t rowi = (size_t)rt * 16 + r + 8 * g;
      res[rowi * H_ + col] = acc[t][r] + bias + hidden[rowi * H_ + col];
    }
  }
}

// ---------------------------------------------------------------------------
// Kernel 6: in-place layernorm over H=1024
// ---------------------------------------------------------------------------
__global__ __launch_bounds__(256) void k_layernorm(
    float* __restrict__ res, const float* __restrict__ gamma, const float* __restrict__ beta)
{
  const int tid = threadIdx.x;
  float* p = res + (size_t)blockIdx.x * H_;
  float4 x = reinterpret_cast<float4*>(p)[tid];
  __shared__ float red[256];

  red[tid] = x.x + x.y + x.z + x.w; __syncthreads();
  for (int s = 128; s > 0; s >>= 1) { if (tid < s) red[tid] += red[tid + s]; __syncthreads(); }
  const float mu = red[0] * (1.0f / H_); __syncthreads();

  const float dx = x.x - mu, dy = x.y - mu, dz = x.z - mu, dw = x.w - mu;
  red[tid] = dx * dx + dy * dy + dz * dz + dw * dw; __syncthreads();
  for (int s = 128; s > 0; s >>= 1) { if (tid < s) red[tid] += red[tid + s]; __syncthreads(); }
  const float inv = 1.0f / sqrtf(red[0] * (1.0f / H_) + 1e-12f);

  const int c = tid * 4;
  x.x = dx * inv * gamma[c + 0] + beta[c + 0];
  x.y = dy * inv * gamma[c + 1] + beta[c + 1];
  x.z = dz * inv * gamma[c + 2] + beta[c + 2];
  x.w = dw * inv * gamma[c + 3] + beta[c + 3];
  reinterpret_cast<float4*>(p)[tid] = x;
}

// ---------------------------------------------------------------------------
extern "C" void kernel_launch(void* const* d_in, const int* in_sizes, int n_in,
                              void* d_out, int out_size, void* d_ws, size_t ws_size,
                              hipStream_t stream) {
  (void)in_sizes; (void)n_in; (void)out_size; (void)ws_size;

  const float* hidden = (const float*)d_in[0];
  const float* Wq  = (const float*)d_in[1];  const float* bq = (const float*)d_in[2];
  const float* Wk  = (const float*)d_in[3];  const float* bk = (const float*)d_in[4];
  const float* Wv  = (const float*)d_in[5];  const float* bv = (const float*)d_in[6];
  const float* Uqr = (const float*)d_in[7];  const float* Uqi = (const float*)d_in[8];
  const float* Ukr = (const float*)d_in[9];  const float* Uki = (const float*)d_in[10];
  const float* Uvr = (const float*)d_in[11]; const float* Uvi = (const float*)d_in[12];
  const float* Wo  = (const float*)d_in[13]; const float* bo  = (const float*)d_in[14];
  const float* gamma = (const float*)d_in[15];
  const float* beta  = (const float*)d_in[16];

  float* out   = (float*)d_out;
  float* res   = out;                              // [B*S, H]   (normed, in-place LN)
  float* probs = out + (size_t)B_ * S_ * H_;       // [B*NH, S, S]

  // workspace layout (halves)
  const size_t QS = (size_t)B_ * NH_ * S_ * DH_;   // 4,194,304
  _Float16* qr    = (_Float16*)d_ws;
  _Float16* qi    = qr + QS;
  _Float16* kr    = qi + QS;
  _Float16* ki    = kr + QS;
  _Float16* vmagT = ki + QS;                       // [B*NH][DH][S]
  _Float16* ctx   = vmagT + QS;                    // [B*S][H]  (same element count)
  _Float16* WoT   = ctx + QS;                      // [H][H]
  _Float16* WqT   = WoT + (size_t)H_ * H_;         // [2*DH][DH]
  _Float16* WkT   = WqT + 2 * DH_ * DH_;
  _Float16* WvT   = WkT + 2 * DH_ * DH_;
  _Float16* UqrT  = WvT + 2 * DH_ * DH_;           // [DH][DH] x6
  _Float16* UqiT  = UqrT + DH_ * DH_;
  _Float16* UkrT  = UqiT + DH_ * DH_;
  _Float16* UkiT  = UkrT + DH_ * DH_;
  _Float16* UvrT  = UkiT + DH_ * DH_;
  _Float16* UviT  = UvrT + DH_ * DH_;

  k_pack<<<(H_ * H_) / 256, 256, 0, stream>>>(
      Wq, Wk, Wv, Uqr, Uqi, Ukr, Uki, Uvr, Uvi, Wo,
      WqT, WkT, WvT, UqrT, UqiT, UkrT, UkiT, UvrT, UviT, WoT);

  k_prep_gate<<<B_ * NH_ * (S_ / 16), 32, 0, stream>>>(
      hidden, bq, bk, bv, WqT, WkT, WvT,
      UqrT, UqiT, UkrT, UkiT, UvrT, UviT,
      qr, qi, kr, ki, vmagT);

  k_scores<<<B_ * NH_ * (S_ / 16) * (S_ / 128), 32, 0, stream>>>(qr, qi, kr, ki, probs);

  k_softmax<<<B_ * NH_ * S_, 256, 0, stream>>>(probs);

  k_ctx<<<B_ * NH_ * (S_ / 16), 32, 0, stream>>>(probs, vmagT, ctx);

  k_outproj<<<(B_ * S_ / 16) * (H_ / 64), 32, 0, stream>>>(ctx, WoT, bo, hidden, res);

  k_layernorm<<<B_ * S_, 256, 0, stream>>>(res, gamma, beta);
}
